// DiscourseAct_32444182954214
// MI455X (gfx1250) — matrix operand order repaired and checked by reading, not compiled, not run on hardware
//
#include <hip/hip_runtime.h>
#include <hip/hip_bf16.h>

// ---- problem constants -----------------------------------------------------
#define SEQ    262144
#define NROWS  (SEQ - 1)        // 262143 output rows
#define H2     256
#define K1     512
#define ODIM   43

// ---- tiling ---------------------------------------------------------------
#define TM     64               // rows per block (4 M-tiles of 16)
#define APAD   520              // A row pitch in halves (512 + 8) -> conflict-free frags
#define PPAD   264              // pre row pitch in halves (256 + 8)
#define BPAD   40               // W1 k-slice row pitch in halves (32 + 8)
#define W2PAD  264              // W2 LDS row pitch in halves (256 + 8)

typedef __attribute__((ext_vector_type(16))) _Float16     v16h;
typedef __attribute__((ext_vector_type(8)))  _Float16     v8h;
typedef __attribute__((ext_vector_type(8)))  float        v8f;
typedef __attribute__((ext_vector_type(4)))  float        v4f;
typedef __attribute__((ext_vector_type(4)))  unsigned int v4u;
typedef __attribute__((ext_vector_type(8)))  int          v8i;
typedef __attribute__((ext_vector_type(4)))  int          v4i;

#if defined(__has_builtin)
#if __has_builtin(__builtin_amdgcn_tensor_load_to_lds) && __has_builtin(__builtin_amdgcn_s_wait_tensorcnt)
#define HAVE_TDM 1
#endif
#endif

__device__ __forceinline__ v16h make_v16h(v8h a, v8h b) {
    v16h r;
#pragma unroll
    for (int i = 0; i < 8; ++i) { r[i] = a[i]; r[i + 8] = b[i]; }
    return r;
}

// tanh via HW transcendentals: tanh(x) = (t-1)/(t+1), t = exp2(2*log2e*x)
__device__ __forceinline__ float fast_tanh(float x) {
    x = fminf(9.0f, fmaxf(-9.0f, x));                       // avoid inf/inf
    float t = __builtin_amdgcn_exp2f(x * 2.885390081777927f);
    return (t - 1.0f) * __builtin_amdgcn_rcpf(t + 1.0f);
}

#ifdef HAVE_TDM
// TDM: contiguous 16KB W1 k-slice -> LDS with HW padding 16B per 64B (BPAD=40 layout)
// 6-arg builtin form (clang-23 / therock-10.0 headers).
__device__ __forceinline__ void tdm_load_w1_slice(const _Float16* gsrc, unsigned lds_byte_off) {
    unsigned long long ga = (unsigned long long)(uintptr_t)gsrc;
    v4u g0;
    g0[0] = 1u;                                       // count=1 (valid), is_restore=0
    g0[1] = lds_byte_off;                             // lds_addr (bytes)
    g0[2] = (unsigned)(ga & 0xffffffffull);           // global_addr[31:0]
    g0[3] = (unsigned)((ga >> 32) & 0x1ffffffull)     // global_addr[56:32]
          | (2u << 30);                               // type = 2 ("image")
    v8i g1;
    g1[0] = (int)((1u << 16)      // data_size = 2 bytes
                | (1u << 20)      // pad_enable
                | (3u << 22)      // pad_interval: 16 DWORDs (64B)
                | (3u << 25));    // pad_amount:    4 DWORDs (16B)
    g1[1] = (int)(0x2000u << 16); // tensor_dim0[15:0] = 8192 at bits[63:48]
    g1[2] = (int)(1u << 16);      // tensor_dim0[31:16]=0 ; tensor_dim1 = 1
    g1[3] = (int)(0x2000u << 16); // tensor_dim1[31:16]=0 ; tile_dim0 = 8192
    g1[4] = 0;                    // tile_dim1 = 0 (1D), tile_dim2 = 0
    g1[5] = 8192;                 // tensor_dim0_stride[31:0]
    g1[6] = 0;
    g1[7] = 0;                    // tensor_dim1_stride = 0
    v4i g2; g2[0] = 0; g2[1] = 0; g2[2] = 0; g2[3] = 0;   // dims 2/3 unused
    v4i g3; g3[0] = 0; g3[1] = 0; g3[2] = 0; g3[3] = 0;   // dim 4 unused
    v8i g4; g4[0] = 0; g4[1] = 0; g4[2] = 0; g4[3] = 0;
            g4[4] = 0; g4[5] = 0; g4[6] = 0; g4[7] = 0;   // extra group (unused)
    __builtin_amdgcn_tensor_load_to_lds(g0, g1, g2, g3, g4, 0);
}
#endif

// ---- prep: W1 f32 -> f16 (k-slice-major), W2 f32 -> f16 padded to 48 rows --
__global__ __launch_bounds__(256) void da_prep_kernel(
    const float* __restrict__ W1, const float* __restrict__ W2,
    _Float16* __restrict__ w1h, _Float16* __restrict__ w2h)
{
    int idx = blockIdx.x * 256 + threadIdx.x;
    if (idx < H2 * K1) {                    // 131072 W1 elements
        int n = idx >> 9, k = idx & 511;    // W1[n][k]
        // slice-major: [ks][n][kk], ks = k/32, contiguous 16KB per slice
        w1h[(k >> 5) * (H2 * 32) + n * 32 + (k & 31)] = (_Float16)W1[idx];
    } else if (idx < H2 * K1 + 48 * H2) {   // 12288 padded W2 elements
        int i2  = idx - H2 * K1;
        int row = i2 >> 8, col = i2 & 255;  // [48][256]
        w2h[i2] = (row < ODIM) ? (_Float16)W2[row * H2 + col] : (_Float16)0.f;
    }
}

// ---- main fused kernel -----------------------------------------------------
__global__ __launch_bounds__(256) void da_mlp_kernel(
    const float* __restrict__ x,        // [SEQ,256]
    const int*   __restrict__ parents,  // [SEQ]
    const float* __restrict__ b1,       // [256]
    const float* __restrict__ b2,       // [43]
    const _Float16* __restrict__ w1h,   // slice-major f16 W1
    const _Float16* __restrict__ w2h,   // [48][256] f16 W2 (padded)
    float* __restrict__ out)            // [NROWS,43]
{
    extern __shared__ char smem[];
    _Float16* sA  = (_Float16*)smem;              // [64][APAD] ; reused as pre [64][PPAD]
    _Float16* sB  = sA + TM * APAD;               // [2][256][BPAD] double-buffered W1 slice
    _Float16* sW2 = sB + 2 * H2 * BPAD;           // [48][W2PAD]
    float*    sb1 = (float*)(sW2 + 48 * W2PAD);   // [256]
    float*    sb2 = sb1 + H2;                     // [48]

    const unsigned SB_OFF = (unsigned)(TM * APAD * sizeof(_Float16));   // sB byte off in LDS
    const unsigned SB_BUF = (unsigned)(H2 * BPAD * sizeof(_Float16));   // one buffer (20480B)

    const int  tid  = threadIdx.x;
    const int  w    = tid >> 5;        // wave id 0..7
    const int  lane = tid & 31;
    const int  l    = lane & 15;       // M / N / K-row index within fragment
    const int  hi   = lane >> 4;       // half-wave selector
    const long base = (long)blockIdx.x * TM;

#ifdef HAVE_TDM
    if (tid < 32) tdm_load_w1_slice(w1h, SB_OFF);          // slice 0 -> buf 0
#endif

    // ---- stage A tile: 64 rows x [child(256) | parent(256)] f32 -> f16 ----
#pragma unroll
    for (int i = 0; i < 16; ++i) {
        int  hr = w + i * 8;           // 128 half-row tasks, wave-uniform
        int  r  = hr >> 1, h = hr & 1;
        long g  = base + r;
        v8h  vh;
#pragma unroll
        for (int e = 0; e < 8; ++e) vh[e] = (_Float16)0.f;
        if (g < NROWS) {
            long srow = (h == 0) ? (g + 1) : (long)parents[g + 1];
            const v4f* p = (const v4f*)(x + srow * H2 + lane * 8);
            v4f f0 = p[0], f1 = p[1];
#pragma unroll
            for (int e = 0; e < 4; ++e) { vh[e] = (_Float16)f0[e]; vh[e + 4] = (_Float16)f1[e]; }
        }
        *(v8h*)(sA + r * APAD + h * H2 + lane * 8) = vh;
    }

    // ---- stage W2 into LDS (48x256 f16) ----
#pragma unroll
    for (int i = 0; i < 6; ++i) {
        int idx = i * 256 + tid;                   // 1536 v8h chunks
        int row = idx >> 5, c8 = (idx & 31) * 8;
        *(v8h*)(sW2 + row * W2PAD + c8) = *(const v8h*)(w2h + row * H2 + c8);
    }
    sb1[tid] = b1[tid];
    if (tid < 48) sb2[tid] = (tid < ODIM) ? b2[tid] : 0.f;

#ifndef HAVE_TDM
    v8h w1a, w1b;
    {
        const v8h* p = (const v8h*)(w1h + tid * 32);
        w1a = p[0]; w1b = p[1];
        _Float16* q = sB + tid * BPAD;
        *(v8h*)q = w1a; *(v8h*)(q + 8) = w1b;
    }
#else
    if (tid < 32) __builtin_amdgcn_s_wait_tensorcnt(0);
#endif
    __syncthreads();

    // ---- GEMM-1: pre[64,256] = A[64,512] x W1^T, f16 WMMA, f32 acc --------
    v8f acc[4][2];
#pragma unroll
    for (int m = 0; m < 4; ++m)
#pragma unroll
        for (int j = 0; j < 2; ++j)
#pragma unroll
            for (int e = 0; e < 8; ++e) acc[m][j][e] = 0.f;

    for (int ks = 0; ks < 16; ++ks) {
#ifdef HAVE_TDM
        if (ks < 15 && tid < 32)       // slice ks+1 -> other buffer (readers retired)
            tdm_load_w1_slice(w1h + (size_t)(ks + 1) * (H2 * 32),
                              SB_OFF + (unsigned)((ks + 1) & 1) * SB_BUF);
#else
        if (ks < 15) {                 // prefetch next slice to regs
            const v8h* p = (const v8h*)(w1h + (size_t)(ks + 1) * (H2 * 32) + tid * 32);
            w1a = p[0]; w1b = p[1];
        }
#endif
        const _Float16* sBb = sB + (ks & 1) * (H2 * BPAD);

        v16h Bf[2];
#pragma unroll
        for (int j = 0; j < 2; ++j) {               // N-tiles 2w, 2w+1
            const _Float16* bp = sBb + ((2 * w + j) * 16 + l) * BPAD + hi * 8;
            Bf[j] = make_v16h(*(const v8h*)bp, *(const v8h*)(bp + 16));
        }
#pragma unroll
        for (int m = 0; m < 4; ++m) {
            const _Float16* ap = sA + (m * 16 + l) * APAD + ks * 32 + hi * 8;
            v16h Af = make_v16h(*(const v8h*)ap, *(const v8h*)(ap + 16));
#pragma unroll
            for (int j = 0; j < 2; ++j)
                acc[m][j] = __builtin_amdgcn_wmma_f32_16x16x32_f16(
                    false, Af, false, Bf[j], (short)0, acc[m][j], false, false);
        }
        if (ks < 15) {
#ifdef HAVE_TDM
            if (tid < 32) __builtin_amdgcn_s_wait_tensorcnt(0);
#else
            _Float16* q = sB + ((ks + 1) & 1) * (H2 * BPAD) + tid * BPAD;
            *(v8h*)q = w1a; *(v8h*)(q + 8) = w1b;
#endif
            __syncthreads();                        // one barrier per k-step
        }
    }

    // ---- epilogue 1: +b1, tanh, store pre (f16) over the A region ---------
    __syncthreads();                                // everyone done reading sA/sB
    _Float16* pre = sA;
#pragma unroll
    for (int m = 0; m < 4; ++m)
#pragma unroll
        for (int j = 0; j < 2; ++j) {
            int   col  = (2 * w + j) * 16 + l;
            float bias = sb1[col];
#pragma unroll
            for (int r = 0; r < 8; ++r) {
                float v = fast_tanh(acc[m][j][r] + bias);   // C/D: row = r + 8*hi
                pre[(m * 16 + r + 8 * hi) * PPAD + col] = (_Float16)v;
            }
        }
    __syncthreads();

    // ---- GEMM-2: out[64,43] = pre[64,256] x W2^T --------------------------
    if (w < 4) {                                    // wave-uniform branch
        v8f acc2[3];
#pragma unroll
        for (int nt = 0; nt < 3; ++nt)
#pragma unroll
            for (int e = 0; e < 8; ++e) acc2[nt][e] = 0.f;

#pragma unroll
        for (int ks = 0; ks < 8; ++ks) {
            const _Float16* ap = pre + (w * 16 + l) * PPAD + ks * 32 + hi * 8;
            v16h Af = make_v16h(*(const v8h*)ap, *(const v8h*)(ap + 16));
#pragma unroll
            for (int nt = 0; nt < 3; ++nt) {
                const _Float16* bp = sW2 + (nt * 16 + l) * W2PAD + ks * 32 + hi * 8;
                v16h Bf = make_v16h(*(const v8h*)bp, *(const v8h*)(bp + 16));
                acc2[nt] = __builtin_amdgcn_wmma_f32_16x16x32_f16(
                    false, Af, false, Bf, (short)0, acc2[nt], false, false);
            }
        }
        // epilogue 2: +b2, guarded stores (divergence only after all WMMAs)
#pragma unroll
        for (int nt = 0; nt < 3; ++nt) {
            int c = nt * 16 + l;
            if (c < ODIM) {
                float bias = sb2[c];
#pragma unroll
                for (int r = 0; r < 8; ++r) {
                    long g = base + w * 16 + r + 8 * hi;
                    if (g < NROWS) out[g * ODIM + c] = acc2[nt][r] + bias;
                }
            }
        }
    }
}

// ---- host glue -------------------------------------------------------------
extern "C" void kernel_launch(void* const* d_in, const int* in_sizes, int n_in,
                              void* d_out, int out_size, void* d_ws, size_t ws_size,
                              hipStream_t stream) {
    const float* x       = (const float*)d_in[0];   // [1,SEQ,256] f32
    const int*   parents = (const int*)  d_in[1];   // [SEQ] int
    const float* W1      = (const float*)d_in[2];   // [256,512] f32
    const float* b1      = (const float*)d_in[3];   // [256] f32
    const float* W2      = (const float*)d_in[4];   // [43,256] f32
    const float* b2      = (const float*)d_in[5];   // [43] f32
    float*       out     = (float*)d_out;           // [NROWS,43] f32

    _Float16* w1h = (_Float16*)d_ws;                // 131072 halves (256 KB)
    _Float16* w2h = w1h + H2 * K1;                  //  12288 halves ( 24 KB)

    {   // prep: 131072 + 12288 elements
        int total  = H2 * K1 + 48 * H2;
        int blocks = (total + 255) / 256;
        da_prep_kernel<<<blocks, 256, 0, stream>>>(W1, W2, w1h, w2h);
    }
    {
        int    blocks = (NROWS + TM - 1) / TM;      // 4096
        size_t shmem  = (size_t)(TM * APAD + 2 * H2 * BPAD + 48 * W2PAD) * sizeof(_Float16)
                      + (size_t)(H2 + 48) * sizeof(float);   // 134,080 B
        da_mlp_kernel<<<blocks, 256, shmem, stream>>>(x, parents, b1, b2, w1h, w2h, out);
    }
}
